// Gemma4FullAttention_47115791237727
// MI455X (gfx1250) — matrix operand order repaired and checked by reading, not compiled, not run on hardware
//
#include <hip/hip_runtime.h>
#include <cstdint>
#include <cstddef>

// ---------------------------------------------------------------------------
// Types for CDNA5 WMMA (wave32): v16bf = 8 VGPRs of bf16, v8f = 8 VGPRs f32
// ---------------------------------------------------------------------------
typedef __attribute__((ext_vector_type(16))) __bf16 v16bf;
typedef __attribute__((ext_vector_type(8)))  __bf16 v8bf;
typedef __attribute__((ext_vector_type(8)))  float  v8f;
typedef __attribute__((ext_vector_type(4)))  unsigned int v4u;
typedef __attribute__((ext_vector_type(8)))  int v8i;
typedef __attribute__((ext_vector_type(4)))  int v4i;

union V16 { v16bf v; v8bf h[2]; };

#define HIDDEN    2048
#define NUM_HEADS 8
#define NUM_KVH   4
#define HEAD_DIM  256
#define SEQ_L     2048
#define BATCH     2
#define M_TOTAL   (BATCH * SEQ_L)   // 4096 tokens

#define MTILE     32                // proj rows per block
#define KSTAGE    128               // proj k-chunk staged in LDS per TDM load

__device__ __forceinline__ unsigned short f32_to_bf16(float f) {
  union { float f; unsigned u; } cv; cv.f = f;
  unsigned u = cv.u;
  unsigned r = ((u & 0x7F800000u) == 0x7F800000u) ? u
                                                  : (u + 0x7FFFu + ((u >> 16) & 1u));
  return (unsigned short)(r >> 16);
}

__device__ __forceinline__ v8f wmma_bf16(v16bf a, v16bf b, v8f c) {
  // (neg_a, A, neg_b, B, c_mod, C, reuse_a, reuse_b)
  return __builtin_amdgcn_wmma_f32_16x16x32_bf16(false, a, false, b, (short)0, c,
                                                 false, false);
}

// ---------------------------------------------------------------------------
// TDM: stage a [mrows x kelems] bf16 tile (row stride K elems) into LDS.
// D# packing per CDNA5 ISA (2-D tensor, groups
// 2/3 zero). data_size=1 -> 2-byte elements. type=2 ("image").
// This toolchain exposes the 6-arg builtin:
//   (uint32x4 g0, int32x8 g1, int32x4 g2, int32x4 g3, int32x8 g4, i32 cpol)
// ---------------------------------------------------------------------------
#if __has_builtin(__builtin_amdgcn_tensor_load_to_lds)
#define HAVE_TDM 1
__device__ __forceinline__ void tdm_load_tile(const unsigned short* gsrc,
                                              unsigned lds_off,
                                              int kelems, int mrows, int K) {
  unsigned long long ga = (unsigned long long)(uintptr_t)gsrc;
  v4u g0;
  g0[0] = 1u;                                   // count=1, user descriptor
  g0[1] = lds_off;                              // LDS byte address
  g0[2] = (unsigned)(ga & 0xFFFFFFFFu);         // global_addr[31:0]
  g0[3] = (unsigned)((ga >> 32) & 0x1FFFFFFu)   // global_addr[56:32]
          | (2u << 30);                         // type=2
  v8i g1;
  g1[0] = 1 << 16;                              // data_size=1 (2 bytes)
  g1[1] = (K & 0xFFFF) << 16;                   // tensor_dim0[15:0]
  g1[2] = ((K >> 16) & 0xFFFF)                  // tensor_dim0[31:16]
          | ((M_TOTAL & 0xFFFF) << 16);         // tensor_dim1[15:0]
  g1[3] = ((M_TOTAL >> 16) & 0xFFFF)            // tensor_dim1[31:16]
          | ((kelems & 0xFFFF) << 16);          // tile_dim0
  g1[4] = mrows & 0xFFFF;                       // tile_dim1 (tile_dim2=0)
  g1[5] = K;                                    // tensor_dim0_stride[31:0]
  g1[6] = 0;                                    // stride0 hi / stride1 lo
  g1[7] = 0;                                    // stride1 hi
  v4i gz4 = {0, 0, 0, 0};
  v8i gz8 = {0, 0, 0, 0, 0, 0, 0, 0};
  __builtin_amdgcn_tensor_load_to_lds(g0, g1, gz4, gz4, gz8, 0);
}
#else
#define HAVE_TDM 0
#endif

// ---------------------------------------------------------------------------
// fp32 -> bf16 cast
// ---------------------------------------------------------------------------
__global__ void cast_f32_to_bf16(const float* __restrict__ in,
                                 unsigned short* __restrict__ out, int n) {
  int i = blockIdx.x * blockDim.x + threadIdx.x;
  int stride = gridDim.x * blockDim.x;
  for (; i < n; i += stride) out[i] = f32_to_bf16(in[i]);
}

// ---------------------------------------------------------------------------
// Projection GEMM: out[m, n] = sum_k A[m,k] * W[n,k]   (A: [M,K], W: [N,K] bf16)
// Block: 256 threads = 8 waves; block computes a 32 x 256 slab (one "head").
// A tile (32 x KSTAGE) staged into LDS by the Tensor Data Mover; wave w owns
// columns [slab + w*32, +32) as 2 n-tiles x 2 m-tiles = 4 WMMA accumulators.
// NORM=1: fused per-row RMSNorm over the 256-wide slab (+ weight + scale).
// OUT_MODE: 0 = bf16 row-major, 1 = bf16 transposed [n][m], 2 = f32 row-major
// ---------------------------------------------------------------------------
template <int NORM, int OUT_MODE>
__global__ __launch_bounds__(256) void proj_kernel(
    const unsigned short* __restrict__ A,
    const unsigned short* __restrict__ W,
    const float* __restrict__ normw,
    void* __restrict__ outp,
    int K, int ldo, float scale) {
  const int lane  = threadIdx.x & 31;
  const int wave  = threadIdx.x >> 5;
  const int hi    = lane >> 4;
  const int ln    = lane & 15;
  const int mbase = blockIdx.x * MTILE;
  const int ncol0 = blockIdx.y * 256 + wave * 32;

  __shared__ unsigned short ldsA[MTILE * KSTAGE];  // 8 KB
  __shared__ float rowsq[MTILE];

  const unsigned short* w0 = W + (size_t)(ncol0 + ln) * K + hi * 16;
  const unsigned short* w1 = W + (size_t)(ncol0 + 16 + ln) * K + hi * 16;

  v8f acc[2][2];
  v8f zf = {};
  acc[0][0] = zf; acc[0][1] = zf; acc[1][0] = zf; acc[1][1] = zf;

  for (int k0 = 0; k0 < K; k0 += KSTAGE) {
    __syncthreads();  // previous stage's LDS reads complete
#if HAVE_TDM
    if (wave == 0) {
      tdm_load_tile(A + (size_t)mbase * K + k0,
                    (unsigned)(uintptr_t)&ldsA[0], KSTAGE, MTILE, K);
      __builtin_amdgcn_s_wait_tensorcnt(0);
    }
#else
    {  // cooperative fallback: 8 KB / 256 threads = two b128 per thread
      const unsigned short* src = A + (size_t)mbase * K + k0;
      for (int t = threadIdx.x; t < MTILE * KSTAGE / 8; t += 256) {
        int r = t / (KSTAGE / 8), c = (t % (KSTAGE / 8)) * 8;
        *(v8bf*)&ldsA[r * KSTAGE + c] = *(const v8bf*)(src + (size_t)r * K + c);
      }
    }
#endif
    __syncthreads();  // publish staged tile to all waves

#pragma unroll
    for (int kk = 0; kk < KSTAGE; kk += 32) {
      V16 a0, a1;
      const unsigned short* l0 = ldsA + ln * KSTAGE + kk;
      const unsigned short* l1 = ldsA + (16 + ln) * KSTAGE + kk;
      a0.h[0] = *(const v8bf*)(l0 + hi * 16);
      a0.h[1] = *(const v8bf*)(l0 + hi * 16 + 8);
      a1.h[0] = *(const v8bf*)(l1 + hi * 16);
      a1.h[1] = *(const v8bf*)(l1 + hi * 16 + 8);
      v16bf b0 = *(const v16bf*)(w0 + k0 + kk);
      v16bf b1 = *(const v16bf*)(w1 + k0 + kk);
      acc[0][0] = wmma_bf16(a0.v, b0, acc[0][0]);
      acc[0][1] = wmma_bf16(a0.v, b1, acc[0][1]);
      acc[1][0] = wmma_bf16(a1.v, b0, acc[1][0]);
      acc[1][1] = wmma_bf16(a1.v, b1, acc[1][1]);
    }
  }

  // Fused RMSNorm over the 256-wide head slab (row-wise), via LDS reduction.
  float rms[2][8];
  if (NORM) {
    __syncthreads();
    if (threadIdx.x < MTILE) rowsq[threadIdx.x] = 0.0f;
    __syncthreads();
#pragma unroll
    for (int mt = 0; mt < 2; ++mt)
#pragma unroll
      for (int e = 0; e < 8; ++e)
        atomicAdd(&rowsq[mt * 16 + e + hi * 8],
                  acc[mt][0][e] * acc[mt][0][e] + acc[mt][1][e] * acc[mt][1][e]);
    __syncthreads();
#pragma unroll
    for (int mt = 0; mt < 2; ++mt)
#pragma unroll
      for (int e = 0; e < 8; ++e)
        rms[mt][e] = rsqrtf(rowsq[mt * 16 + e + hi * 8] * (1.0f / 256.0f) + 1e-6f);
  }

  const int c0 = wave * 32 + ln;  // column within the 256-wide head
  const int c1 = c0 + 16;
#pragma unroll
  for (int mt = 0; mt < 2; ++mt) {
#pragma unroll
    for (int e = 0; e < 8; ++e) {
      const int row = mbase + mt * 16 + e + hi * 8;
      float v0 = acc[mt][0][e], v1 = acc[mt][1][e];
      if (NORM) {
        v0 *= rms[mt][e] * normw[c0] * scale;
        v1 *= rms[mt][e] * normw[c1] * scale;
      }
      if (OUT_MODE == 0) {
        unsigned short* o = (unsigned short*)outp;
        o[(size_t)row * ldo + ncol0 + ln]      = f32_to_bf16(v0);
        o[(size_t)row * ldo + ncol0 + 16 + ln] = f32_to_bf16(v1);
      } else if (OUT_MODE == 1) {  // transposed: out[n][m], ldo = M_TOTAL
        unsigned short* o = (unsigned short*)outp;
        o[(size_t)(ncol0 + ln) * ldo + row]      = f32_to_bf16(v0);
        o[(size_t)(ncol0 + 16 + ln) * ldo + row] = f32_to_bf16(v1);
      } else {  // f32 row-major
        float* o = (float*)outp;
        o[(size_t)row * ldo + ncol0 + ln]      = v0;
        o[(size_t)row * ldo + ncol0 + 16 + ln] = v1;
      }
    }
  }
}

// ---------------------------------------------------------------------------
// Causal GQA flash attention. One wave per (batch, head, 16-query tile).
// Q: [M_TOTAL, 2048] bf16 (pre-normalized, pre-scaled by d^-1/2 * log2(e))
// K: [M_TOTAL, 1024] bf16 (pre-normalized)
// V: transposed [NUM_KVH*256][M_TOTAL] bf16 -> contiguous along keys
// O: [M_TOTAL, 2048] bf16
// ---------------------------------------------------------------------------
__global__ __launch_bounds__(32) void attn_kernel(
    const unsigned short* __restrict__ Q,
    const unsigned short* __restrict__ Kb,
    const unsigned short* __restrict__ Vt,
    unsigned short* __restrict__ O) {
  const int lane  = threadIdx.x & 31;
  const int hi    = lane >> 4;
  const int ln    = lane & 15;
  const int qt    = blockIdx.x;
  const int h     = blockIdx.y;
  const int b     = blockIdx.z;
  const int qbase = qt * 16;
  const int kvh   = h >> 1;  // GQA: n_rep = 2
  const int mq    = b * SEQ_L + qbase;

  __shared__ unsigned short ldsP[16 * 32];

  // Preload Q fragments for all 8 d-steps (d = 256 = 8 * 32)
  V16 qf[8];
  const unsigned short* qrow =
      Q + (size_t)(mq + ln) * (NUM_HEADS * HEAD_DIM) + h * HEAD_DIM;
#pragma unroll
  for (int d = 0; d < 8; ++d) {
    qf[d].h[0] = *(const v8bf*)(qrow + d * 32 + hi * 16);
    qf[d].h[1] = *(const v8bf*)(qrow + d * 32 + hi * 16 + 8);
  }

  v8f zf = {};
  v8f oacc[16];
#pragma unroll
  for (int t = 0; t < 16; ++t) oacc[t] = zf;

  float mrow[8], lrow[8];
#pragma unroll
  for (int e = 0; e < 8; ++e) { mrow[e] = -INFINITY; lrow[e] = 0.0f; }

  const int nkb = (qbase + 16 + 31) / 32;  // 32-key blocks (causal bound)
  for (int kb = 0; kb < nkb; ++kb) {
    const int k0 = kb * 32;
    // ---- S = Q @ K^T for two 16-key tiles ----
    v8f s0 = {};
    v8f s1 = {};
    const unsigned short* kr0 = Kb +
        (size_t)(b * SEQ_L + k0 + ln) * (NUM_KVH * HEAD_DIM) + kvh * HEAD_DIM + hi * 16;
    const unsigned short* kr1 = kr0 + (size_t)16 * (NUM_KVH * HEAD_DIM);
#pragma unroll
    for (int d = 0; d < 8; ++d) {
      v16bf b0 = *(const v16bf*)(kr0 + d * 32);
      v16bf b1 = *(const v16bf*)(kr1 + d * 32);
      s0 = wmma_bf16(qf[d].v, b0, s0);
      s1 = wmma_bf16(qf[d].v, b1, s1);
    }
    // ---- causal mask ----
#pragma unroll
    for (int e = 0; e < 8; ++e) {
      const int qi = qbase + e + hi * 8;
      if (k0 + ln > qi)      s0[e] = -INFINITY;
      if (k0 + 16 + ln > qi) s1[e] = -INFINITY;
    }
    // ---- online softmax (exp2; q pre-scaled by log2e * d^-1/2) ----
#pragma unroll
    for (int e = 0; e < 8; ++e) {
      float mx = fmaxf(s0[e], s1[e]);
      for (int off = 1; off < 16; off <<= 1)
        mx = fmaxf(mx, __shfl_xor(mx, off, 16));
      const float mn    = fmaxf(mrow[e], mx);
      const float alpha = exp2f(mrow[e] - mn);
      const float p0    = exp2f(s0[e] - mn);
      const float p1    = exp2f(s1[e] - mn);
      float rs = p0 + p1;
      for (int off = 1; off < 16; off <<= 1)
        rs += __shfl_xor(rs, off, 16);
      lrow[e] = lrow[e] * alpha + rs;
      mrow[e] = mn;
#pragma unroll
      for (int t = 0; t < 16; ++t) oacc[t][e] *= alpha;
      // stash P (C-layout) into LDS so it can be re-read in A-fragment order
      ldsP[(e + hi * 8) * 32 + ln]      = f32_to_bf16(p0);
      ldsP[(e + hi * 8) * 32 + 16 + ln] = f32_to_bf16(p1);
    }
    __syncthreads();
    // ---- reload P as a 16x32 A fragment ----
    V16 pa;
    pa.h[0] = *(const v8bf*)(ldsP + ln * 32 + hi * 16);
    pa.h[1] = *(const v8bf*)(ldsP + ln * 32 + hi * 16 + 8);
    // ---- O += P @ V over 16 d-tiles (V transposed: contiguous along keys) ----
#pragma unroll
    for (int dt = 0; dt < 16; ++dt) {
      const unsigned short* vp = Vt +
          (size_t)(kvh * HEAD_DIM + dt * 16 + ln) * M_TOTAL + b * SEQ_L + k0 + hi * 16;
      v16bf vb = *(const v16bf*)vp;
      oacc[dt] = wmma_bf16(pa.v, vb, oacc[dt]);
    }
    __syncthreads();
  }

  // ---- epilogue: normalize by row sum, write bf16 ----
#pragma unroll
  for (int e = 0; e < 8; ++e) {
    const float linv = 1.0f / lrow[e];
    const int row = b * SEQ_L + qbase + e + hi * 8;
    unsigned short* orow = O + (size_t)row * (NUM_HEADS * HEAD_DIM) + h * HEAD_DIM;
#pragma unroll
    for (int dt = 0; dt < 16; ++dt)
      orow[dt * 16 + ln] = f32_to_bf16(oacc[dt][e] * linv);
  }
}

// ---------------------------------------------------------------------------
// Host-side launcher
// ---------------------------------------------------------------------------
extern "C" void kernel_launch(void* const* d_in, const int* in_sizes, int n_in,
                              void* d_out, int out_size, void* d_ws, size_t ws_size,
                              hipStream_t stream) {
  const float* hs  = (const float*)d_in[0];  // [2, 2048, 2048]
  const float* qw  = (const float*)d_in[1];  // [2048, 2048]
  const float* kw  = (const float*)d_in[2];  // [1024, 2048]
  const float* vw  = (const float*)d_in[3];  // [1024, 2048]
  const float* ow  = (const float*)d_in[4];  // [2048, 2048]
  const float* qnw = (const float*)d_in[5];  // [256]
  const float* knw = (const float*)d_in[6];  // [256]

  char* ws = (char*)d_ws;
  size_t off = 0;
  auto alloc = [&](size_t bytes) -> unsigned short* {
    unsigned short* p = (unsigned short*)(ws + off);
    off = (off + bytes + 255) & ~(size_t)255;
    return p;
  };
  unsigned short* x_bf    = alloc((size_t)M_TOTAL * HIDDEN * 2);
  unsigned short* qw_bf   = alloc((size_t)NUM_HEADS * HEAD_DIM * HIDDEN * 2);
  unsigned short* kw_bf   = alloc((size_t)NUM_KVH * HEAD_DIM * HIDDEN * 2);
  unsigned short* vw_bf   = alloc((size_t)NUM_KVH * HEAD_DIM * HIDDEN * 2);
  unsigned short* ow_bf   = alloc((size_t)HIDDEN * NUM_HEADS * HEAD_DIM * 2);
  unsigned short* q_bf    = alloc((size_t)M_TOTAL * NUM_HEADS * HEAD_DIM * 2);
  unsigned short* k_bf    = alloc((size_t)M_TOTAL * NUM_KVH * HEAD_DIM * 2);
  unsigned short* vT_bf   = alloc((size_t)NUM_KVH * HEAD_DIM * M_TOTAL * 2);
  unsigned short* attn_bf = alloc((size_t)M_TOTAL * NUM_HEADS * HEAD_DIM * 2);
  (void)ws_size; (void)n_in; (void)in_sizes; (void)out_size;

  // 1) fp32 -> bf16 casts
  auto cast = [&](const float* src, unsigned short* dst, int n) {
    cast_f32_to_bf16<<<2048, 256, 0, stream>>>(src, dst, n);
  };
  cast(hs, x_bf, M_TOTAL * HIDDEN);
  cast(qw, qw_bf, NUM_HEADS * HEAD_DIM * HIDDEN);
  cast(kw, kw_bf, NUM_KVH * HEAD_DIM * HIDDEN);
  cast(vw, vw_bf, NUM_KVH * HEAD_DIM * HIDDEN);
  cast(ow, ow_bf, HIDDEN * NUM_HEADS * HEAD_DIM);

  // 2) Projections (fold softmax scale * log2(e) into normalized q)
  const float qscale = 0.0625f * 1.44269504088896f;  // 256^-0.5 * log2(e)
  dim3 blk(256);
  proj_kernel<1, 0><<<dim3(M_TOTAL / MTILE, NUM_HEADS), blk, 0, stream>>>(
      x_bf, qw_bf, qnw, q_bf, HIDDEN, NUM_HEADS * HEAD_DIM, qscale);
  proj_kernel<1, 0><<<dim3(M_TOTAL / MTILE, NUM_KVH), blk, 0, stream>>>(
      x_bf, kw_bf, knw, k_bf, HIDDEN, NUM_KVH * HEAD_DIM, 1.0f);
  proj_kernel<0, 1><<<dim3(M_TOTAL / MTILE, NUM_KVH), blk, 0, stream>>>(
      x_bf, vw_bf, nullptr, vT_bf, HIDDEN, M_TOTAL, 1.0f);

  // 3) Causal GQA flash attention: one wave per (b, head, 16-query tile)
  attn_kernel<<<dim3(SEQ_L / 16, NUM_HEADS, BATCH), dim3(32), 0, stream>>>(
      q_bf, k_bf, vT_bf, attn_bf);

  // 4) O projection -> f32 output
  proj_kernel<0, 2><<<dim3(M_TOTAL / MTILE, HIDDEN / 256), blk, 0, stream>>>(
      attn_bf, ow_bf, nullptr, d_out, NUM_HEADS * HEAD_DIM, HIDDEN, 1.0f);
}
